// model_FABind_layer_20418274525617
// MI455X (gfx1250) — compile-verified, matrix operand order checked by reading, not run on hardware
//
#include <hip/hip_runtime.h>
#include <math.h>

// ---------------- problem constants (match reference) ----------------
#define B_    16
#define Nc_   64
#define Np_   1000
#define H_    128
#define L_    2
#define E_    600000
#define S_    (2 + Nc_ + Np_)      // 1066
#define NTOT  (B_ * S_)            // 17056
#define FC_   56
#define FP_   1280
#define SCALE_ 5.0f
#define MTILE 32                   // edges per block (2 WMMA M-tiles)

typedef __attribute__((ext_vector_type(16))) __bf16 v16bf;
typedef __attribute__((ext_vector_type(8)))  float  v8f;

__device__ __forceinline__ unsigned short f2bf(float f) {
  unsigned u = __float_as_uint(f);
  u += 0x7FFFu + ((u >> 16) & 1u);           // round-to-nearest-even
  return (unsigned short)(u >> 16);
}
__device__ __forceinline__ float silu_f(float v) { return v / (1.0f + __expf(-v)); }

// ---------------- generic zero ----------------
__global__ void k_zero(float* p, int n) {
  int i = blockIdx.x * blockDim.x + threadIdx.x;
  if (i < n) p[i] = 0.0f;
}

// ---------------- weight packing: [K,N] fp32 row-major -> bf16 WMMA B-fragments
// frag index t = (nt*KT + kt)*32 + lane ; each lane owns 16 contiguous bf16:
//   lane<16 : n = nt*16+lane,     k = kt*32 + i
//   lane>=16: n = nt*16+lane-16,  k = kt*32 + 16 + i
__global__ void k_pack(const float* __restrict__ W, int K, int N, unsigned short* __restrict__ out) {
  int t = blockIdx.x * blockDim.x + threadIdx.x;
  int KT = K >> 5, NT = N >> 4;
  int total = NT * KT * 32;
  if (t >= total) return;
  int lane = t & 31;
  int kt = (t >> 5) % KT;
  int nt = (t >> 5) / KT;
  int n = nt * 16 + (lane & 15);
  int kbase = kt * 32 + ((lane >> 4) << 4);
  unsigned short* o = out + (size_t)t * 16;
#pragma unroll
  for (int i = 0; i < 16; ++i) o[i] = f2bf(W[(size_t)(kbase + i) * N + n]);
}

// ---------------- initial embedding: h[N,H], h_bf mirror, x = coords/SCALE
__global__ __launch_bounds__(128) void k_embed(
    const float* __restrict__ cf, const float* __restrict__ pf,
    const float* __restrict__ coords,
    const float* __restrict__ glb_c, const float* __restrict__ glb_p,
    const float* __restrict__ Wc, const float* __restrict__ bc,
    const float* __restrict__ Wp, const float* __restrict__ bp,
    float* __restrict__ h, unsigned short* __restrict__ hbf, float* __restrict__ x) {
  int n = blockIdx.x, j = threadIdx.x;
  __shared__ float feat[FP_];
  int s = n % S_, b = n / S_;
  float hv;
  if (s == 0) {
    hv = glb_c[j];
  } else if (s == Nc_ + 1) {
    hv = glb_p[j];
  } else if (s <= Nc_) {
    int row = b * Nc_ + (s - 1);
    if (j < FC_) feat[j] = cf[(size_t)row * FC_ + j];
    __syncthreads();
    float acc = bc[j];
    for (int k = 0; k < FC_; ++k) acc += feat[k] * Wc[k * H_ + j];
    hv = acc;
  } else {
    int row = b * Np_ + (s - Nc_ - 2);
    for (int k = j; k < FP_; k += H_) feat[k] = pf[(size_t)row * FP_ + k];
    __syncthreads();
    float acc = bp[j];
    for (int k = 0; k < FP_; ++k) acc += feat[k] * Wp[k * H_ + j];
    hv = acc;
  }
  h[(size_t)n * H_ + j] = hv;
  hbf[(size_t)n * H_ + j] = f2bf(hv);
  if (j < 3) x[n * 3 + j] = coords[n * 3 + j] * (1.0f / SCALE_);
}

// ---------------- in-degree count ----------------
__global__ void k_cnt(const int* __restrict__ dst, float* __restrict__ cnt, int E) {
  int i = blockIdx.x * blockDim.x + threadIdx.x;
  if (i < E) atomicAdd(&cnt[dst[i]], 1.0f);
}

// ---------------- fused per-edge MLP chain (WMMA bf16), 32 edges / block ----
// 8 waves = 8 N-tiles; each wave computes TWO 16x16 output tiles (M-tiles),
// reusing every B (weight) fragment for 2 WMMAs -> halves L2 weight traffic.
__global__ __launch_bounds__(256) void k_edge(
    const unsigned short* __restrict__ hbf, const float* __restrict__ x,
    const int* __restrict__ src, const int* __restrict__ dst,
    const unsigned short* __restrict__ W1p, const float* __restrict__ w1last,
    const float* __restrict__ b1,
    const unsigned short* __restrict__ W2p, const float* __restrict__ b2,
    const unsigned short* __restrict__ Wx1p, const float* __restrict__ bx1,
    const float* __restrict__ wx2, const float* __restrict__ bx2p,
    float* __restrict__ x_acc, float* __restrict__ m_agg) {
  __shared__ __attribute__((aligned(64))) unsigned short eh[MTILE][256];  // [h[dst]|h[src]] bf16
  __shared__ __attribute__((aligned(64))) unsigned short ab[MTILE][128];  // a1 then m (bf16)
  __shared__ float a3s[MTILE][128];
  __shared__ float d2s[MTILE];
  __shared__ float dif[MTILE][3];
  __shared__ int   dN[MTILE];
  __shared__ int   sN[MTILE];
  __shared__ float wE[MTILE];

  const int tid  = threadIdx.x;
  const int lane = tid & 31;
  const int wv   = tid >> 5;      // wave = N-tile (0..7)
  const int e0   = blockIdx.x * MTILE;

  if (tid < MTILE) {
    int e = e0 + tid;
    int sI = src[e], dI = dst[e];
    sN[tid] = sI; dN[tid] = dI;
    float dx = x[dI * 3 + 0] - x[sI * 3 + 0];
    float dy = x[dI * 3 + 1] - x[sI * 3 + 1];
    float dz = x[dI * 3 + 2] - x[sI * 3 + 2];
    dif[tid][0] = dx; dif[tid][1] = dy; dif[tid][2] = dz;
    d2s[tid] = dx * dx + dy * dy + dz * dz;
  }
  __syncthreads();
  // gather bf16 h rows: 16 chunks (32B) per edge, 512 tasks over 256 threads
#pragma unroll
  for (int it = 0; it < 2; ++it) {
    int idx = tid + it * 256;
    int r = idx >> 4, c = idx & 15;
    int node = (c < 8) ? dN[r] : sN[r];
    int colb = (c < 8) ? c * 16 : (c - 8) * 16;
    const uint4* p = (const uint4*)(hbf + (size_t)node * H_ + colb);
    uint4 v0 = p[0], v1 = p[1];
    uint4* q = (uint4*)&eh[r][c * 16];
    q[0] = v0; q[1] = v1;
  }
  __syncthreads();

  const int arow  = lane & 15;
  const int ksel  = (lane >> 4) << 4;
  const int col   = wv * 16 + arow;
  const int rb    = (lane >> 4) * 8;

  // ---- GEMM1: e_in[:,0:256] @ We_w1[0:256] ----
  v8f acc0 = {0.f, 0.f, 0.f, 0.f, 0.f, 0.f, 0.f, 0.f};
  v8f acc1 = {0.f, 0.f, 0.f, 0.f, 0.f, 0.f, 0.f, 0.f};
  const v16bf* W1f = (const v16bf*)W1p;
#pragma unroll
  for (int kt = 0; kt < 8; ++kt) {
    v16bf b  = W1f[(wv * 8 + kt) * 32 + lane];
    v16bf a0 = *(const v16bf*)&eh[arow][kt * 32 + ksel];
    v16bf a1 = *(const v16bf*)&eh[16 + arow][kt * 32 + ksel];
    acc0 = __builtin_amdgcn_wmma_f32_16x16x32_bf16(false, a0, false, b, (short)0, acc0, false, false);
    acc1 = __builtin_amdgcn_wmma_f32_16x16x32_bf16(false, a1, false, b, (short)0, acc1, false, false);
  }
  {  // + bias + d2 * We_w1[256,:]  (rank-1, fp32) ; SiLU ; stash bf16
    float bias = b1[col], wl = w1last[col];
#pragma unroll
    for (int v = 0; v < 8; ++v) {
      int r0 = rb + v, r1 = 16 + rb + v;
      float t0 = acc0[v] + bias + d2s[r0] * wl;
      float t1 = acc1[v] + bias + d2s[r1] * wl;
      ab[r0][col] = f2bf(silu_f(t0));
      ab[r1][col] = f2bf(silu_f(t1));
    }
  }
  __syncthreads();

  // ---- GEMM2: a1 @ We_w2 -> m ----
  acc0 = (v8f){0.f, 0.f, 0.f, 0.f, 0.f, 0.f, 0.f, 0.f};
  acc1 = (v8f){0.f, 0.f, 0.f, 0.f, 0.f, 0.f, 0.f, 0.f};
  const v16bf* W2f = (const v16bf*)W2p;
#pragma unroll
  for (int kt = 0; kt < 4; ++kt) {
    v16bf b  = W2f[(wv * 4 + kt) * 32 + lane];
    v16bf a0 = *(const v16bf*)&ab[arow][kt * 32 + ksel];
    v16bf a1 = *(const v16bf*)&ab[16 + arow][kt * 32 + ksel];
    acc0 = __builtin_amdgcn_wmma_f32_16x16x32_bf16(false, a0, false, b, (short)0, acc0, false, false);
    acc1 = __builtin_amdgcn_wmma_f32_16x16x32_bf16(false, a1, false, b, (short)0, acc1, false, false);
  }
  float m0[8], m1[8];
  {
    float bias = b2[col];
#pragma unroll
    for (int v = 0; v < 8; ++v) {
      m0[v] = silu_f(acc0[v] + bias);
      m1[v] = silu_f(acc1[v] + bias);
    }
  }
  __syncthreads();  // all GEMM2 reads of ab done
#pragma unroll
  for (int v = 0; v < 8; ++v) {
    int r0 = rb + v, r1 = 16 + rb + v;
    ab[r0][col] = f2bf(m0[v]);
    ab[r1][col] = f2bf(m1[v]);
    atomicAdd(&m_agg[(size_t)dN[r0] * H_ + col], m0[v]);  // agg = seg(m, dst)
    atomicAdd(&m_agg[(size_t)dN[r1] * H_ + col], m1[v]);
  }
  __syncthreads();

  // ---- GEMM3: m @ Wx_w1 -> a3 ----
  acc0 = (v8f){0.f, 0.f, 0.f, 0.f, 0.f, 0.f, 0.f, 0.f};
  acc1 = (v8f){0.f, 0.f, 0.f, 0.f, 0.f, 0.f, 0.f, 0.f};
  const v16bf* Wx1f = (const v16bf*)Wx1p;
#pragma unroll
  for (int kt = 0; kt < 4; ++kt) {
    v16bf b  = Wx1f[(wv * 4 + kt) * 32 + lane];
    v16bf a0 = *(const v16bf*)&ab[arow][kt * 32 + ksel];
    v16bf a1 = *(const v16bf*)&ab[16 + arow][kt * 32 + ksel];
    acc0 = __builtin_amdgcn_wmma_f32_16x16x32_bf16(false, a0, false, b, (short)0, acc0, false, false);
    acc1 = __builtin_amdgcn_wmma_f32_16x16x32_bf16(false, a1, false, b, (short)0, acc1, false, false);
  }
  {
    float bias = bx1[col];
#pragma unroll
    for (int v = 0; v < 8; ++v) {
      a3s[rb + v][col]      = silu_f(acc0[v] + bias);
      a3s[16 + rb + v][col] = silu_f(acc1[v] + bias);
    }
  }
  __syncthreads();

  // ---- w = a3 @ Wx_w2 + bx2 ; scatter diff*w ----
  if (tid < MTILE) {
    float s = bx2p[0];
    for (int k = 0; k < H_; ++k) s += a3s[tid][k] * wx2[k];
    wE[tid] = s;
  }
  __syncthreads();
  if (tid < 3 * MTILE) {
    int r = tid / 3, k = tid % 3;
    atomicAdd(&x_acc[(size_t)dN[r] * 3 + k], dif[r][k] * wE[r]);
  }
}

// ---------------- per-node update: x += acc/cnt ; h += MLP([h,agg]) --------
__global__ __launch_bounds__(128) void k_node(
    float* __restrict__ h, unsigned short* __restrict__ hbf, float* __restrict__ x,
    const float* __restrict__ x_acc, const float* __restrict__ cnt,
    const float* __restrict__ m_agg,
    const float* __restrict__ Wh1, const float* __restrict__ bh1,
    const float* __restrict__ Wh2, const float* __restrict__ bh2) {
  int n = blockIdx.x, j = threadIdx.x;
  __shared__ float hin[2 * H_];
  __shared__ float ts[H_];
  hin[j]       = h[(size_t)n * H_ + j];
  hin[H_ + j]  = m_agg[(size_t)n * H_ + j];
  if (j < 3) {
    float c = cnt[n];
    c = (c < 1.0f) ? 1.0f : c;
    x[n * 3 + j] += x_acc[n * 3 + j] / c;
  }
  __syncthreads();
  float t = bh1[j];
  for (int k = 0; k < 2 * H_; ++k) t += hin[k] * Wh1[k * H_ + j];
  ts[j] = silu_f(t);
  __syncthreads();
  float u = bh2[j];
  for (int k = 0; k < H_; ++k) u += ts[k] * Wh2[k * H_ + j];
  float hv = hin[j] + u;
  h[(size_t)n * H_ + j]   = hv;
  hbf[(size_t)n * H_ + j] = f2bf(hv);
}

// ---------------- compound center per batch + copy ccm to output ----------
__global__ __launch_bounds__(64) void k_center(const float* __restrict__ x,
                                               const float* __restrict__ ccm,
                                               float* __restrict__ centers,
                                               float* __restrict__ out) {
  int b = blockIdx.x, t = threadIdx.x;
  __shared__ float sm[64][3];
  int n = b * S_ + 1 + t;
  sm[t][0] = x[n * 3 + 0] * SCALE_;
  sm[t][1] = x[n * 3 + 1] * SCALE_;
  sm[t][2] = x[n * 3 + 2] * SCALE_;
  __syncthreads();
  for (int off = 32; off > 0; off >>= 1) {
    if (t < off) {
      sm[t][0] += sm[t + off][0];
      sm[t][1] += sm[t + off][1];
      sm[t][2] += sm[t + off][2];
    }
    __syncthreads();
  }
  if (t < 3) {
    float c = sm[0][t] * (1.0f / (float)Nc_);
    centers[b * 3 + t] = c;
    out[b * 3 + t] = c;                    // compound_center_pred
    out[48 + b * 3 + t] = ccm[b * 3 + t];  // compound_coords_mean (passthrough)
  }
}

// ---------------- clipped distances -----------------------------------
__global__ void k_dist(const float* __restrict__ pn, const float* __restrict__ ccm,
                       const float* __restrict__ centers, float* __restrict__ out) {
  int i = blockIdx.x * blockDim.x + threadIdx.x;
  if (i >= B_ * Np_) return;
  int b = i / Np_;
  float px = pn[i * 3 + 0], py = pn[i * 3 + 1], pz = pn[i * 3 + 2];
  float dx = px - centers[b * 3 + 0], dy = py - centers[b * 3 + 1], dz = pz - centers[b * 3 + 2];
  float dp = sqrtf(dx * dx + dy * dy + dz * dz);
  dp = fminf(dp, 10.0f);
  dx = px - ccm[b * 3 + 0]; dy = py - ccm[b * 3 + 1]; dz = pz - ccm[b * 3 + 2];
  float dt = sqrtf(dx * dx + dy * dy + dz * dz);
  dt = fminf(dt, 10.0f);
  out[96 + i] = dp;
  out[96 + B_ * Np_ + i] = dt;
}

// =====================================================================
extern "C" void kernel_launch(void* const* d_in, const int* in_sizes, int n_in,
                              void* d_out, int out_size, void* d_ws, size_t ws_size,
                              hipStream_t stream) {
  (void)in_sizes; (void)n_in; (void)out_size; (void)ws_size;
  const float* cf     = (const float*)d_in[0];
  const float* pf     = (const float*)d_in[1];
  const float* coords = (const float*)d_in[2];
  const float* pnxyz  = (const float*)d_in[3];
  const float* ccm    = (const float*)d_in[4];
  const float* glb_c  = (const float*)d_in[5];
  const float* glb_p  = (const float*)d_in[6];
  const float* Wc     = (const float*)d_in[7];
  const float* bc     = (const float*)d_in[8];
  const float* Wp     = (const float*)d_in[9];
  const float* bp     = (const float*)d_in[10];
  const float* We_w1  = (const float*)d_in[11];
  const float* We_b1  = (const float*)d_in[12];
  const float* We_w2  = (const float*)d_in[13];
  const float* We_b2  = (const float*)d_in[14];
  const float* Wx_w1  = (const float*)d_in[15];
  const float* Wx_b1  = (const float*)d_in[16];
  const float* Wx_w2  = (const float*)d_in[17];
  const float* Wx_b2  = (const float*)d_in[18];
  const float* Wh_w1  = (const float*)d_in[19];
  const float* Wh_b1  = (const float*)d_in[20];
  const float* Wh_w2  = (const float*)d_in[21];
  const float* Wh_b2  = (const float*)d_in[22];
  const int*   ei     = (const int*)d_in[23];
  const int* srcI = ei;
  const int* dstI = ei + E_;
  float* out = (float*)d_out;

  // ---- workspace partition (256B aligned) ----
  char* ws = (char*)d_ws;
  size_t off = 0;
  auto alloc = [&](size_t bytes) -> char* {
    char* p = ws + off;
    off = (off + bytes + 255) & ~(size_t)255;
    return p;
  };
  float*          hf      = (float*)alloc((size_t)NTOT * H_ * 4);
  unsigned short* hbf     = (unsigned short*)alloc((size_t)NTOT * H_ * 2);
  float*          xbuf    = (float*)alloc((size_t)NTOT * 3 * 4);
  float*          x_acc   = (float*)alloc((size_t)NTOT * 3 * 4);
  float*          cntb    = (float*)alloc((size_t)NTOT * 4);
  float*          m_agg   = (float*)alloc((size_t)NTOT * H_ * 4);
  float*          centers = (float*)alloc(48 * 4);
  unsigned short* pW1[L_], *pW2[L_], *pWx1[L_];
  for (int l = 0; l < L_; ++l) {
    pW1[l]  = (unsigned short*)alloc((size_t)8 * 8 * 32 * 16 * 2);  // K=256
    pW2[l]  = (unsigned short*)alloc((size_t)8 * 4 * 32 * 16 * 2);  // K=128
    pWx1[l] = (unsigned short*)alloc((size_t)8 * 4 * 32 * 16 * 2);  // K=128
  }

  // ---- pack weights to bf16 WMMA fragments ----
  for (int l = 0; l < L_; ++l) {
    k_pack<<<(2048 + 255) / 256, 256, 0, stream>>>(We_w1 + (size_t)l * 257 * H_, 256, H_, pW1[l]);
    k_pack<<<(1024 + 255) / 256, 256, 0, stream>>>(We_w2 + (size_t)l * H_ * H_, H_, H_, pW2[l]);
    k_pack<<<(1024 + 255) / 256, 256, 0, stream>>>(Wx_w1 + (size_t)l * H_ * H_, H_, H_, pWx1[l]);
  }

  // ---- initial node embeddings + normalized coords ----
  k_embed<<<NTOT, 128, 0, stream>>>(cf, pf, coords, glb_c, glb_p, Wc, bc, Wp, bp,
                                    hf, hbf, xbuf);

  // ---- in-degree counts ----
  k_zero<<<(NTOT + 255) / 256, 256, 0, stream>>>(cntb, NTOT);
  k_cnt<<<(E_ + 255) / 256, 256, 0, stream>>>(dstI, cntb, E_);

  // ---- EGNN layers ----
  for (int l = 0; l < L_; ++l) {
    k_zero<<<(NTOT * 3 + 255) / 256, 256, 0, stream>>>(x_acc, NTOT * 3);
    k_zero<<<(NTOT * H_ + 255) / 256, 256, 0, stream>>>(m_agg, NTOT * H_);
    k_edge<<<E_ / MTILE, 256, 0, stream>>>(
        hbf, xbuf, srcI, dstI,
        pW1[l], We_w1 + (size_t)l * 257 * H_ + 256 * H_, We_b1 + l * H_,
        pW2[l], We_b2 + l * H_,
        pWx1[l], Wx_b1 + l * H_,
        Wx_w2 + (size_t)l * H_, Wx_b2 + l,
        x_acc, m_agg);
    k_node<<<NTOT, 128, 0, stream>>>(
        hf, hbf, xbuf, x_acc, cntb, m_agg,
        Wh_w1 + (size_t)l * 256 * H_, Wh_b1 + l * H_,
        Wh_w2 + (size_t)l * H_ * H_, Wh_b2 + l * H_);
  }

  // ---- outputs ----
  k_center<<<B_, 64, 0, stream>>>(xbuf, ccm, centers, out);
  k_dist<<<(B_ * Np_ + 255) / 256, 256, 0, stream>>>(pnxyz, ccm, centers, out);
}
